// MSACMixer_54305566490740
// MI455X (gfx1250) — matrix operand order repaired and checked by reading, not compiled, not run on hardware
//
#include <hip/hip_runtime.h>
#include <hip/hip_bf16.h>

// ---------------- CDNA5 WMMA types ----------------
typedef __attribute__((ext_vector_type(16))) __bf16 v16bf;
typedef __attribute__((ext_vector_type(8)))  float  v8f;

// ---------------- problem sizes ----------------
#define BATCH   8
#define SEQ     4096
#define DDIM    1024
#define MROWS   (BATCH * SEQ)     // 32768
#define NUP     (2 * DDIM)        // 2048

// ---------------- GEMM tiling ----------------
#define BM 128
#define BN 128
#define BK 32
#define LDSS 40                   // shorts per LDS row (80B, stride-20-banks, conflict-free)
#define NIT  (DDIM / BK)          // 32 K-steps

// ---------------- bf16 helpers ----------------
__device__ __forceinline__ unsigned short f2bf(float f) {
    union { float f; unsigned int u; } v; v.f = f;
    unsigned int r = v.u + 0x7FFFu + ((v.u >> 16) & 1u);   // RNE
    return (unsigned short)(r >> 16);
}
__device__ __forceinline__ float bf2f(unsigned int h16) {
    union { float f; unsigned int u; } v; v.u = (h16 & 0xFFFFu) << 16; return v.f;
}
__device__ __forceinline__ unsigned int pack2(float a, float b) {
    return (unsigned int)f2bf(a) | ((unsigned int)f2bf(b) << 16);
}

// ---------------- CDNA5 async global->LDS copy (ASYNCcnt path) ----------------
// ISA 10.x / 08_async_tensor: INST_OFFSET is added to BOTH the LDS address
// (VDST vgpr) and the global address (VADDR vgpr pair), so two b128 ops with
// offset:0 / offset:16 copy 32B per lane from one base pair.
__device__ __forceinline__ unsigned lds_off(const void* p) {
    return (unsigned)(unsigned long long)p;     // generic LDS addr[31:0] == LDS offset
}
__device__ __forceinline__ void async_copy_32B(unsigned lds_addr, const void* gaddr) {
    asm volatile("global_load_async_to_lds_b128 %0, %1, off\n\t"
                 "global_load_async_to_lds_b128 %0, %1, off offset:16"
                 :: "v"(lds_addr), "v"(gaddr) : "memory");
}
__device__ __forceinline__ void wait_async0() {
    asm volatile("s_wait_asynccnt 0x0" ::: "memory");
}

// ---------------- LDS fragment loaders ----------------
// A frag (16x32 bf16): lane l -> row (l&15); lanes 0-15 K={0..7,16..23}, lanes 16-31 K={8..15,24..31}
__device__ __forceinline__ v16bf load_fragA(const unsigned short* s, int row, int c) {
    union { uint4 u[2]; v16bf v; } f;
    f.u[0] = *(const uint4*)(s + row * LDSS + c * 8);
    f.u[1] = *(const uint4*)(s + row * LDSS + c * 8 + 16);
    return f.v;
}
// B frag (32x16 bf16), B staged as [N][K]: lane l -> col (l&15); lanes 0-15 K={0..7,16..23}, lanes 16-31 K={8..15,24..31}
__device__ __forceinline__ v16bf load_fragB(const unsigned short* s, int row, int c) {
    union { uint4 u[2]; v16bf v; } f;
    f.u[0] = *(const uint4*)(s + row * LDSS + c * 8);
    f.u[1] = *(const uint4*)(s + row * LDSS + c * 8 + 16);
    return f.v;
}

// ---------------- kernel 1: fp32 -> bf16 weight conversion ----------------
__global__ void cvt_bf16_kernel(const float* __restrict__ src,
                                unsigned short* __restrict__ dst, int n) {
    int i = blockIdx.x * blockDim.x + threadIdx.x;
    if (i < n) dst[i] = f2bf(src[i]);
}

// ---------------- kernel 2: GEMM1  gv = x @ w_up^T, split sigmoid/val epilogue ----------------
// x: fp32 [M,1024] ; wUp: bf16 [2048,1024] ([N][K]-major)
// A path: global fp32 -> VALU convert -> ds_store (needs conversion)
// B path: global_load_async_to_lds_b128 (pure copy)
// Double-buffered LDS.
__global__ __launch_bounds__(256)
void gemm1_kernel(const float* __restrict__ x,
                  const unsigned short* __restrict__ wUp,
                  unsigned short* __restrict__ gateOut,
                  unsigned short* __restrict__ valOut) {
    __shared__ unsigned short As[2][BM * LDSS];
    __shared__ unsigned short Bs[2][BN * LDSS];

    const int tid  = threadIdx.x;
    const int lane = tid & 31;
    const int wid  = tid >> 5;
    const int wm   = (wid & 1) * 64;      // wave M offset in tile
    const int wn   = (wid >> 1) * 32;     // wave N offset in tile
    const int lrow = lane & 15;
    const int lc   = lane >> 4;

    const int m0 = blockIdx.x * BM;
    const int n0 = blockIdx.y * BN;

    // staging assignment: 256 threads cover 128 rows x 2 halves (16 elems each)
    const int grow = tid >> 1;
    const int gcol = (tid & 1) << 4;      // 0 or 16 (elements)
    const float*          ap = x   + (size_t)(m0 + grow) * DDIM + gcol;
    const unsigned short* bp = wUp + (size_t)(n0 + grow) * DDIM + gcol;
    const int sofs = grow * LDSS + gcol;

    v8f acc[4][2];
#pragma unroll
    for (int i = 0; i < 4; ++i)
#pragma unroll
        for (int j = 0; j < 2; ++j) acc[i][j] = {};

    // ---- prologue: stage K-tile 0 into buffer 0 ----
    {
        float4 a0 = *(const float4*)(ap + 0);
        float4 a1 = *(const float4*)(ap + 4);
        float4 a2 = *(const float4*)(ap + 8);
        float4 a3 = *(const float4*)(ap + 12);
        async_copy_32B(lds_off(&Bs[0][sofs]), bp);
        uint4 pa0, pa1;
        pa0.x = pack2(a0.x, a0.y); pa0.y = pack2(a0.z, a0.w);
        pa0.z = pack2(a1.x, a1.y); pa0.w = pack2(a1.z, a1.w);
        pa1.x = pack2(a2.x, a2.y); pa1.y = pack2(a2.z, a2.w);
        pa1.z = pack2(a3.x, a3.y); pa1.w = pack2(a3.z, a3.w);
        *(uint4*)(&As[0][sofs])     = pa0;
        *(uint4*)(&As[0][sofs] + 8) = pa1;
    }
    wait_async0();
    __syncthreads();

    for (int it = 0; it < NIT; ++it) {
        const int cur = it & 1;
        const unsigned short* Ac = As[cur];
        const unsigned short* Bc = Bs[cur];

        // ---- stage next tile into the other buffer (overlaps with compute) ----
        if (it + 1 < NIT) {
            const int kn = (it + 1) * BK;
            async_copy_32B(lds_off(&Bs[cur ^ 1][sofs]), bp + kn);
            float4 a0 = *(const float4*)(ap + kn + 0);
            float4 a1 = *(const float4*)(ap + kn + 4);
            float4 a2 = *(const float4*)(ap + kn + 8);
            float4 a3 = *(const float4*)(ap + kn + 12);
            if (it + 2 < NIT) __builtin_prefetch(ap + kn + BK, 0, 3);
            uint4 pa0, pa1;
            pa0.x = pack2(a0.x, a0.y); pa0.y = pack2(a0.z, a0.w);
            pa0.z = pack2(a1.x, a1.y); pa0.w = pack2(a1.z, a1.w);
            pa1.x = pack2(a2.x, a2.y); pa1.y = pack2(a2.z, a2.w);
            pa1.z = pack2(a3.x, a3.y); pa1.w = pack2(a3.z, a3.w);
            unsigned short* An = As[cur ^ 1];
            *(uint4*)(&An[sofs])     = pa0;
            *(uint4*)(&An[sofs] + 8) = pa1;
        }

        // ---- compute on current buffer ----
        v16bf af[4], bf[2];
#pragma unroll
        for (int i = 0; i < 4; ++i) af[i] = load_fragA(Ac, wm + i * 16 + lrow, lc);
#pragma unroll
        for (int j = 0; j < 2; ++j) bf[j] = load_fragB(Bc, wn + j * 16 + lrow, lc);
#pragma unroll
        for (int i = 0; i < 4; ++i)
#pragma unroll
            for (int j = 0; j < 2; ++j)
                acc[i][j] = __builtin_amdgcn_wmma_f32_16x16x32_bf16(
                    false, af[i], false, bf[j], (short)0, acc[i][j], false, false);

        wait_async0();
        __syncthreads();
    }

    // epilogue: C layout -> m_local = r + 8*(lane>>4), n_local = lane&15
    const int mbase = m0 + wm + (lc << 3);
    const int nbase = n0 + wn + lrow;
    if (n0 < DDIM) {            // block-uniform: sigmoid -> gate
#pragma unroll
        for (int i = 0; i < 4; ++i)
#pragma unroll
            for (int j = 0; j < 2; ++j)
#pragma unroll
                for (int r = 0; r < 8; ++r) {
                    const int m = mbase + i * 16 + r;
                    const int n = nbase + j * 16;
                    float g = 1.0f / (1.0f + __expf(-acc[i][j][r]));
                    gateOut[(size_t)m * DDIM + n] = f2bf(g);
                }
    } else {                    // raw val
#pragma unroll
        for (int i = 0; i < 4; ++i)
#pragma unroll
            for (int j = 0; j < 2; ++j)
#pragma unroll
                for (int r = 0; r < 8; ++r) {
                    const int m = mbase + i * 16 + r;
                    const int n = nbase + j * 16;
                    valOut[(size_t)m * DDIM + (n - DDIM)] = f2bf(acc[i][j][r]);
                }
    }
}

// ---------------- kernel 3: per-row softmax mixing weights ----------------
__global__ __launch_bounds__(256)
void scale_gate_kernel(const float* __restrict__ x,
                       const float* __restrict__ w_gate,
                       float* __restrict__ sw) {
    const int row  = blockIdx.x * 8 + (threadIdx.x >> 5);
    const int lane = threadIdx.x & 31;
    const float* xr = x + (size_t)row * DDIM;
    float a0 = 0.f, a1 = 0.f, a2 = 0.f;
    for (int d = lane; d < DDIM; d += 32) {
        float xv = xr[d];
        a0 = fmaf(xv, w_gate[d], a0);
        a1 = fmaf(xv, w_gate[DDIM + d], a1);
        a2 = fmaf(xv, w_gate[2 * DDIM + d], a2);
    }
#pragma unroll
    for (int off = 16; off > 0; off >>= 1) {
        a0 += __shfl_down(a0, off, 32);
        a1 += __shfl_down(a1, off, 32);
        a2 += __shfl_down(a2, off, 32);
    }
    if (lane == 0) {
        float m  = fmaxf(a0, fmaxf(a1, a2));
        float e0 = __expf(a0 - m), e1 = __expf(a1 - m), e2 = __expf(a2 - m);
        float inv = 1.0f / (e0 + e1 + e2);
        sw[row * 3 + 0] = e0 * inv;
        sw[row * 3 + 1] = e1 * inv;
        sw[row * 3 + 2] = e2 * inv;
    }
}

// ---------------- kernel 4: causal depthwise convs + softmax mix + gate ----------------
// fused[s,d] = gate[s,d] * sum_{tau=0..6} c[tau,d] * val[s-6+tau, d]
// c[tau,d] = sw2*w7[d,tau] + (tau>=2)*sw1*w5[d,tau-2] + (tau>=4)*sw0*w3[d,tau-4]
__global__ __launch_bounds__(256)
void conv_fuse_kernel(const unsigned short* __restrict__ val,
                      const unsigned short* __restrict__ gate,
                      const float* __restrict__ sw,
                      const float* __restrict__ w3,
                      const float* __restrict__ w5,
                      const float* __restrict__ w7,
                      unsigned short* __restrict__ fused) {
    const size_t idx = (size_t)blockIdx.x * blockDim.x + threadIdx.x; // over M * D/2
    const int dp = (int)(idx & 511);          // d-pair index (D/2 = 512)
    const int s  = (int)(idx >> 9);
    const int t  = s & (SEQ - 1);             // position within batch
    const int d0 = dp << 1;

    const float s0 = sw[s * 3 + 0], s1 = sw[s * 3 + 1], s2 = sw[s * 3 + 2];

    float acc0 = 0.f, acc1 = 0.f;
#pragma unroll
    for (int j = 0; j < 7; ++j) {
        float c0 = s2 * w7[(d0 + 0) * 7 + j];
        float c1 = s2 * w7[(d0 + 1) * 7 + j];
        if (j >= 2) { c0 = fmaf(s1, w5[(d0 + 0) * 5 + (j - 2)], c0);
                      c1 = fmaf(s1, w5[(d0 + 1) * 5 + (j - 2)], c1); }
        if (j >= 4) { c0 = fmaf(s0, w3[(d0 + 0) * 3 + (j - 4)], c0);
                      c1 = fmaf(s0, w3[(d0 + 1) * 3 + (j - 4)], c1); }
        const int ts = t - 6 + j;
        if (ts >= 0) {
            unsigned int p = *(const unsigned int*)(val + ((size_t)(s - 6 + j) * DDIM + d0));
            acc0 = fmaf(c0, bf2f(p), acc0);
            acc1 = fmaf(c1, bf2f(p >> 16), acc1);
        }
    }
    unsigned int gp = *(const unsigned int*)(gate + (size_t)s * DDIM + d0);
    acc0 *= bf2f(gp);
    acc1 *= bf2f(gp >> 16);
    *(unsigned int*)(fused + (size_t)s * DDIM + d0) = pack2(acc0, acc1);
}

// ---------------- kernel 5: GEMM2  out = fused @ w_down^T (fp32 out) ----------------
// fused: bf16 [M,1024] ; wDn: bf16 [1024,1024] ([N][K]-major)
// Both A and B tiles staged via async global->LDS DMA, double-buffered.
__global__ __launch_bounds__(256)
void gemm2_kernel(const unsigned short* __restrict__ fused,
                  const unsigned short* __restrict__ wDn,
                  float* __restrict__ out) {
    __shared__ unsigned short As[2][BM * LDSS];
    __shared__ unsigned short Bs[2][BN * LDSS];

    const int tid  = threadIdx.x;
    const int lane = tid & 31;
    const int wid  = tid >> 5;
    const int wm   = (wid & 1) * 64;
    const int wn   = (wid >> 1) * 32;
    const int lrow = lane & 15;
    const int lc   = lane >> 4;

    const int m0 = blockIdx.x * BM;
    const int n0 = blockIdx.y * BN;

    const int grow = tid >> 1;
    const int gcol = (tid & 1) << 4;
    const unsigned short* ap = fused + (size_t)(m0 + grow) * DDIM + gcol;
    const unsigned short* bp = wDn  + (size_t)(n0 + grow) * DDIM + gcol;
    const int sofs = grow * LDSS + gcol;

    v8f acc[4][2];
#pragma unroll
    for (int i = 0; i < 4; ++i)
#pragma unroll
        for (int j = 0; j < 2; ++j) acc[i][j] = {};

    // prologue: async-stage tile 0 into buffer 0
    async_copy_32B(lds_off(&As[0][sofs]), ap);
    async_copy_32B(lds_off(&Bs[0][sofs]), bp);
    wait_async0();
    __syncthreads();

    for (int it = 0; it < NIT; ++it) {
        const int cur = it & 1;
        const unsigned short* Ac = As[cur];
        const unsigned short* Bc = Bs[cur];

        if (it + 1 < NIT) {
            const int kn = (it + 1) * BK;
            async_copy_32B(lds_off(&As[cur ^ 1][sofs]), ap + kn);
            async_copy_32B(lds_off(&Bs[cur ^ 1][sofs]), bp + kn);
        }

        v16bf af[4], bf[2];
#pragma unroll
        for (int i = 0; i < 4; ++i) af[i] = load_fragA(Ac, wm + i * 16 + lrow, lc);
#pragma unroll
        for (int j = 0; j < 2; ++j) bf[j] = load_fragB(Bc, wn + j * 16 + lrow, lc);
#pragma unroll
        for (int i = 0; i < 4; ++i)
#pragma unroll
            for (int j = 0; j < 2; ++j)
                acc[i][j] = __builtin_amdgcn_wmma_f32_16x16x32_bf16(
                    false, af[i], false, bf[j], (short)0, acc[i][j], false, false);

        wait_async0();
        __syncthreads();
    }

    const int mbase = m0 + wm + (lc << 3);
    const int nbase = n0 + wn + lrow;
#pragma unroll
    for (int i = 0; i < 4; ++i)
#pragma unroll
        for (int j = 0; j < 2; ++j)
#pragma unroll
            for (int r = 0; r < 8; ++r) {
                const int m = mbase + i * 16 + r;
                const int n = nbase + j * 16;
                out[(size_t)m * DDIM + n] = acc[i][j][r];
            }
}

// ---------------- launcher ----------------
extern "C" void kernel_launch(void* const* d_in, const int* in_sizes, int n_in,
                              void* d_out, int out_size, void* d_ws, size_t ws_size,
                              hipStream_t stream) {
    const float* x      = (const float*)d_in[0];   // [8,4096,1024]
    const float* w_up   = (const float*)d_in[1];   // [2048,1024]
    const float* w_down = (const float*)d_in[2];   // [1024,1024]
    const float* w_gate = (const float*)d_in[3];   // [3,1024]
    const float* w3     = (const float*)d_in[4];   // [1024,1,3]
    const float* w5     = (const float*)d_in[5];   // [1024,1,5]
    const float* w7     = (const float*)d_in[6];   // [1024,1,7]
    float* out = (float*)d_out;

    // workspace layout (bytes)
    char* ws = (char*)d_ws;
    unsigned short* wUpB  = (unsigned short*)(ws);                     // 4 MB
    unsigned short* wDnB  = (unsigned short*)(ws + (4ull  << 20));     // 2 MB
    unsigned short* valB  = (unsigned short*)(ws + (6ull  << 20));     // 64 MB
    unsigned short* gateB = (unsigned short*)(ws + (70ull << 20));     // 64 MB
    unsigned short* fusedB= (unsigned short*)(ws + (134ull<< 20));     // 64 MB
    float*          swF   = (float*)         (ws + (198ull<< 20));     // 384 KB

    // 1) weight conversion to bf16
    {
        int n1 = NUP * DDIM;          // 2097152
        int n2 = DDIM * DDIM;         // 1048576
        cvt_bf16_kernel<<<(n1 + 255) / 256, 256, 0, stream>>>(w_up,   wUpB, n1);
        cvt_bf16_kernel<<<(n2 + 255) / 256, 256, 0, stream>>>(w_down, wDnB, n2);
    }
    // 2) softmax mixing weights (independent of gemm1)
    scale_gate_kernel<<<MROWS / 8, 256, 0, stream>>>(x, w_gate, swF);
    // 3) GEMM1 + sigmoid/split epilogue
    {
        dim3 grid(MROWS / BM, NUP / BN);   // 256 x 16
        gemm1_kernel<<<grid, 256, 0, stream>>>(x, wUpB, gateB, valB);
    }
    // 4) conv + mix + gate
    {
        size_t total = (size_t)MROWS * (DDIM / 2);     // 16.7M threads
        conv_fuse_kernel<<<(unsigned)(total / 256), 256, 0, stream>>>(
            valB, gateB, swF, w3, w5, w7, fusedB);
    }
    // 5) GEMM2 -> fp32 output
    {
        dim3 grid(MROWS / BM, DDIM / BN);  // 256 x 8
        gemm2_kernel<<<grid, 256, 0, stream>>>(fusedB, wDnB, out);
    }
    (void)in_sizes; (void)n_in; (void)out_size; (void)ws_size;
}
